// CrossNetwork_91242285237050
// MI455X (gfx1250) — compile-verified, matrix operand order checked by reading, not run on hardware
//
#include <hip/hip_runtime.h>

typedef _Float16 half_t;
typedef __attribute__((ext_vector_type(16))) _Float16 v16h;
typedef __attribute__((ext_vector_type(8)))  _Float16 v8h;
typedef __attribute__((ext_vector_type(4)))  _Float16 v4h;
typedef __attribute__((ext_vector_type(8)))  float    v8f;

#define B_   4096
#define F_   40
#define E_   64
#define T_   2
#define M_   3
#define EPS_ 1e-5f
#define K2E  128          // 2*E
#define FP   48           // F padded to multiple of 16
#define KG   64           // g (source-field) dim padded to 64 for K
#define NBN  (T_*F_*K2E)  // 10240 BatchNorm positions
#define GK   (F_*E_)      // 2560 gate K

// Load one 16x32 f16 WMMA fragment row-slice for this lane.
// Layout convention (CDNA5 16-bit A/B): element h<8 -> K = kb + hi*8 + h,
// h>=8 -> K = kb + 16 + hi*8 + (h-8).  With operand stored [row][K] row-major
// this is exactly two contiguous 16-byte loads.
__device__ __forceinline__ v16h frag_ld(const half_t* __restrict__ row_kb_hi8) {
  v8h lo = *(const v8h*)(row_kb_hi8);
  v8h hi = *(const v8h*)(row_kb_hi8 + 16);
  return __builtin_shufflevector(lo, hi, 0,1,2,3,4,5,6,7,8,9,10,11,12,13,14,15);
}

// ---------------------------------------------------------------------------
// Kernel 1: batch-independent prep.
//  - adjacency: relu(prod_m masker) -> LayerNorm(source dim) -> masked softmax
//    -> store transposed f16, zero-padded to 48x64 (K pad MUST be zero)
//  - W^T  f16 : [T][e=64][k=128]
//  - gate_w^T f16 : [f=48][k=2560]
//  - fold BatchNorm into interleaved per-position (scale, shift) float2
// ---------------------------------------------------------------------------
__global__ __launch_bounds__(256) void prep_kernel(
    const float* __restrict__ masker, const float* __restrict__ ln_g,
    const float* __restrict__ ln_b,   const float* __restrict__ w_t,
    const float* __restrict__ gate_w,
    const float* __restrict__ bn_g, const float* __restrict__ bn_b,
    const float* __restrict__ bn_m, const float* __restrict__ bn_v,
    half_t* __restrict__ adjT, half_t* __restrict__ wT16,
    half_t* __restrict__ gwT16, float2* __restrict__ bnss)
{
  int gid = blockIdx.x * blockDim.x + threadIdx.x;
  int nth = gridDim.x * blockDim.x;

  if (gid < T_ * F_) {                 // one thread per (tower, dest-field f)
    int t = gid / F_, f = gid % F_;
    float p[F_], msk[F_];
    float mu = 0.f;
    for (int g = 0; g < F_; ++g) {
      float v = masker[((t*M_+0)*F_+g)*F_+f]
              * masker[((t*M_+1)*F_+g)*F_+f]
              * masker[((t*M_+2)*F_+g)*F_+f];
      v = v > 0.f ? v : 0.f;           // relu
      msk[g] = (v != 0.f) ? 1.f : 0.f;
      p[g] = v; mu += v;
    }
    mu *= (1.f / F_);
    float var = 0.f;
    for (int g = 0; g < F_; ++g) { float d = p[g] - mu; var += d * d; }
    var *= (1.f / F_);
    float inv = rsqrtf(var + EPS_);
    float mx = -1e30f;
    for (int g = 0; g < F_; ++g) {     // LN (over g) + mask bias + eye
      float a = (p[g] - mu) * inv * ln_g[g] + ln_b[g];
      a += (1.f - msk[g]) * (-1e9f);
      if (g == f) a += 1.f;
      p[g] = a; mx = fmaxf(mx, a);
    }
    float s = 0.f;
    for (int g = 0; g < F_; ++g) { float e = __expf(p[g] - mx); p[g] = e; s += e; }
    float r = 1.f / s;
    for (int g = 0; g < F_; ++g)
      adjT[(t*FP + f)*KG + g] = (half_t)(p[g] * r * msk[g]);
  }
  // zero pad regions of adjT (disjoint from the writes above)
  for (int i = gid; i < T_*FP*KG; i += nth) {
    int rem = i % (FP*KG);
    int f = rem / KG, g = rem % KG;
    if (f >= F_ || g >= F_) adjT[i] = (half_t)0.f;
  }
  // W^T f16 : wT16[t][e][k] = w_t[t][k][e]
  for (int i = gid; i < T_*E_*K2E; i += nth) {
    int t = i / (E_*K2E), rem = i % (E_*K2E);
    int e = rem / K2E, k = rem % K2E;
    wT16[i] = (half_t)w_t[(t*K2E + k)*E_ + e];
  }
  // gate_w^T f16 : gwT16[f][k] = gate_w[k][f]
  for (int i = gid; i < FP*GK; i += nth) {
    int f = i / GK, k = i % GK;
    gwT16[i] = (half_t)((f < F_) ? gate_w[k*F_ + f] : 0.f);
  }
  // fold BN: y = x*scale + shift, interleaved (scale, shift)
  for (int i = gid; i < NBN; i += nth) {
    float sc = bn_g[i] * rsqrtf(bn_v[i] + EPS_);
    bnss[i] = make_float2(sc, bn_b[i] - bn_m[i] * sc);
  }
}

// ---------------------------------------------------------------------------
// Kernel 2: per (2 batches, tower): nei GEMM -> BN -> concat -> main GEMM+bias.
// 256 threads = 8 waves; wave w handles (batch w/4, e-tile w%4); k-tiles outer
// with 3 M-tile accumulators so each B fragment is loaded once.
// ---------------------------------------------------------------------------
__global__ __launch_bounds__(256) void main_kernel(
    const float* __restrict__ x, const float* __restrict__ b_t,
    const half_t* __restrict__ adjT, const half_t* __restrict__ wT16,
    const float2* __restrict__ bnss, half_t* __restrict__ xcur)
{
  __shared__ __align__(16) half_t s_adjT[FP * KG];      // [f][g] zero K-pad
  __shared__ __align__(16) half_t s_xT[2][E_ * KG];     // [e][g] zero K-pad
  __shared__ __align__(16) half_t s_xc[2][FP * K2E];    // [f][k], rows>=40 junk

  int tid = threadIdx.x;
  int t   = blockIdx.x & 1;
  int b0  = (blockIdx.x >> 1) * 2;

  // stage adjT (uint4 copies), x (float4 loads -> xT LDS + BN'd xc first half)
  for (int i = tid; i < (FP*KG)/8; i += 256)
    ((uint4*)s_adjT)[i] = ((const uint4*)(adjT + t*FP*KG))[i];
  for (int i = tid; i < 2*E_*(KG-F_); i += 256) {        // zero g-pad of xT
    int j = i / (E_*(KG-F_)), rem = i % (E_*(KG-F_));
    int e = rem / (KG-F_), g = F_ + rem % (KG-F_);
    s_xT[j][e*KG + g] = (half_t)0.f;
  }
  for (int j = 0; j < 2; ++j) {
    const float* xb = x + (size_t)(b0 + j) * F_ * E_;
    for (int i4 = tid; i4 < (F_*E_)/4; i4 += 256) {
      float4 v = ((const float4*)xb)[i4];
      int i = i4 * 4;
      int f = i >> 6, e = i & 63;
      // transposed f16 copy for the nei B operand
      s_xT[j][(e+0)*KG + f] = (half_t)v.x;
      s_xT[j][(e+1)*KG + f] = (half_t)v.y;
      s_xT[j][(e+2)*KG + f] = (half_t)v.z;
      s_xT[j][(e+3)*KG + f] = (half_t)v.w;
      // BN'd first half of xc (contiguous 8-byte store)
      int k0 = (t*F_ + f)*K2E + e;
      float2 s0 = bnss[k0+0], s1 = bnss[k0+1], s2 = bnss[k0+2], s3 = bnss[k0+3];
      v4h h4 = { (half_t)(v.x*s0.x + s0.y), (half_t)(v.y*s1.x + s1.y),
                 (half_t)(v.z*s2.x + s2.y), (half_t)(v.w*s3.x + s3.y) };
      *(v4h*)&s_xc[j][f*K2E + e] = h4;
    }
  }
  __syncthreads();

  int wave = tid >> 5, lane = tid & 31;
  int bsel = wave >> 2, nt = wave & 3;      // batch, e-tile
  int ln15 = lane & 15;
  int hi8  = ((lane >> 4) & 1) * 8;

  // ---- nei[f,e] = sum_g adjT[f,g] * x[g,e] : A=s_adjT[f][g], B=s_xT[e][g] ----
  {
    v8f acc0 = {}, acc1 = {}, acc2 = {};
    for (int kt = 0; kt < 2; ++kt) {
      int kb = kt * 32;
      v16h bfr = frag_ld(&s_xT[bsel][(nt*16 + ln15)*KG + kb + hi8]);
      v16h a0  = frag_ld(&s_adjT[( 0 + ln15)*KG + kb + hi8]);
      v16h a1  = frag_ld(&s_adjT[(16 + ln15)*KG + kb + hi8]);
      v16h a2  = frag_ld(&s_adjT[(32 + ln15)*KG + kb + hi8]);
      acc0 = __builtin_amdgcn_wmma_f32_16x16x32_f16(false, a0, false, bfr, (short)0, acc0, false, false);
      acc1 = __builtin_amdgcn_wmma_f32_16x16x32_f16(false, a1, false, bfr, (short)0, acc1, false, false);
      acc2 = __builtin_amdgcn_wmma_f32_16x16x32_f16(false, a2, false, bfr, (short)0, acc2, false, false);
    }
    // D: row = mt*16 + v + hi8, col = nt*16 + ln15. Rows >= 40 are dead (junk
    // allowed in s_xc rows 40..47): tiles 0/1 always valid; tile 2 valid rows
    // need a clamped BN index only for hi lanes (row>=40, dead anyway).
    int e = nt*16 + ln15;
    int base = (t*F_)*K2E + 64 + e;
#pragma unroll
    for (int v = 0; v < 8; ++v) {
      int r0 = v + hi8;
      float2 ss;
      ss = bnss[base + (r0     )*K2E];
      s_xc[bsel][( 0 + r0)*K2E + 64 + e] = (half_t)(acc0[v]*ss.x + ss.y);
      ss = bnss[base + (r0 + 16)*K2E];
      s_xc[bsel][(16 + r0)*K2E + 64 + e] = (half_t)(acc1[v]*ss.x + ss.y);
      int rc = hi8 ? 0 : (r0 + 32);                 // clamp dead-row BN index
      ss = bnss[base + rc*K2E];
      s_xc[bsel][(32 + r0)*K2E + 64 + e] = (half_t)(acc2[v]*ss.x + ss.y);
    }
  }
  __syncthreads();

  // ---- xcur[f,e] = sum_k xc[f,k] * W[k,e] + bias : A=s_xc[f][k], B=wT[e][k] ----
  {
    const half_t* wt = wT16 + t * E_ * K2E;
    v8f acc0 = {}, acc1 = {}, acc2 = {};
    for (int kt = 0; kt < 4; ++kt) {
      int kb = kt * 32;
      v16h bfr = frag_ld(&wt[(nt*16 + ln15)*K2E + kb + hi8]);
      v16h a0  = frag_ld(&s_xc[bsel][( 0 + ln15)*K2E + kb + hi8]);
      v16h a1  = frag_ld(&s_xc[bsel][(16 + ln15)*K2E + kb + hi8]);
      v16h a2  = frag_ld(&s_xc[bsel][(32 + ln15)*K2E + kb + hi8]);
      acc0 = __builtin_amdgcn_wmma_f32_16x16x32_f16(false, a0, false, bfr, (short)0, acc0, false, false);
      acc1 = __builtin_amdgcn_wmma_f32_16x16x32_f16(false, a1, false, bfr, (short)0, acc1, false, false);
      acc2 = __builtin_amdgcn_wmma_f32_16x16x32_f16(false, a2, false, bfr, (short)0, acc2, false, false);
    }
    int e = nt*16 + ln15;
    half_t* xo = xcur + ((size_t)(b0 + bsel)*T_ + t)*F_*E_;
    const float* bt = b_t + (t*F_)*E_ + e;
#pragma unroll
    for (int v = 0; v < 8; ++v) {
      int r0 = v + hi8;
      xo[( 0 + r0)*E_ + e] = (half_t)(acc0[v] + bt[( 0 + r0)*E_]);
      xo[(16 + r0)*E_ + e] = (half_t)(acc1[v] + bt[(16 + r0)*E_]);
    }
    if (hi8 == 0) {                       // tile 2: rows 32..39 only (lanes 0-15)
#pragma unroll
      for (int v = 0; v < 8; ++v)
        xo[(32 + v)*E_ + e] = (half_t)(acc2[v] + bt[(32 + v)*E_]);
    }
  }
}

// ---------------------------------------------------------------------------
// Kernel 3: gate GEMM (16 (b,t)-rows x 2560 @ 2560 x 48) + softmax over
// fields + weighted sum -> out (B,T,E). 128 threads = 4 waves (3 do WMMA).
// A = xcur[row][k] (row-major, contiguous), B = gwT[f][k] (contiguous).
// ---------------------------------------------------------------------------
__global__ __launch_bounds__(128) void gate_kernel(
    const half_t* __restrict__ xcur, const half_t* __restrict__ gwT16,
    const float* __restrict__ gate_b, float* __restrict__ out)
{
  __shared__ float s_g[16 * FP];
  __shared__ float s_w[16 * F_];
  int tid  = threadIdx.x;
  int bt0  = blockIdx.x * 16;
  int wave = tid >> 5, lane = tid & 31;
  int ln15 = lane & 15;
  int hi8  = ((lane >> 4) & 1) * 8;

  if (wave < 3) {
    int nt = wave;                          // field-tile (48 = 3 x 16)
    const half_t* ar = xcur + (size_t)(bt0 + ln15) * GK + hi8;
    const half_t* br = gwT16 + (size_t)(nt*16 + ln15) * GK + hi8;
    v8f acc = {};
    for (int kt = 0; kt < 80; ++kt) {       // K = 2560
      int kb = kt * 32;
      __builtin_prefetch(ar + kb + 1024, 0, 1);   // global_prefetch ~2KB ahead
      __builtin_prefetch(br + kb + 1024, 0, 1);
      v16h a = frag_ld(ar + kb);
      v16h b = frag_ld(br + kb);
      acc = __builtin_amdgcn_wmma_f32_16x16x32_f16(false, a, false, b,
                                                   (short)0, acc, false, false);
    }
    int f  = nt*16 + ln15;
    int fc = (f < F_) ? f : 0;
    float gb = gate_b[fc];
#pragma unroll
    for (int v = 0; v < 8; ++v) {
      int m = v + hi8;
      s_g[m*FP + f] = acc[v] + ((f < F_) ? gb : -1e30f);
    }
  }
  __syncthreads();

  if (tid < 16) {                           // softmax over fields, fp32
    int m = tid;
    float mx = -1e30f;
    for (int f = 0; f < F_; ++f) mx = fmaxf(mx, s_g[m*FP + f]);
    float s = 0.f;
    for (int f = 0; f < F_; ++f) { float e = __expf(s_g[m*FP + f] - mx); s_w[m*F_ + f] = e; s += e; }
    float r = 1.f / s;
    for (int f = 0; f < F_; ++f) s_w[m*F_ + f] *= r;
  }
  __syncthreads();

  // out[bt,e] = sum_f w[f]*xcur[f,e]; 4 e's per thread with 8-byte f16 loads
  for (int p = tid; p < 16 * (E_/4); p += 128) {
    int m = p >> 4, e0 = (p & 15) * 4;
    const half_t* xr = xcur + (size_t)(bt0 + m) * GK + e0;
    float a0 = 0.f, a1 = 0.f, a2 = 0.f, a3 = 0.f;
    for (int f = 0; f < F_; ++f) {
      v4h xv = *(const v4h*)(xr + f*E_);
      float wf = s_w[m*F_ + f];
      a0 += wf * (float)xv[0]; a1 += wf * (float)xv[1];
      a2 += wf * (float)xv[2]; a3 += wf * (float)xv[3];
    }
    float4 o = make_float4(a0, a1, a2, a3);
    *(float4*)&out[(size_t)(bt0 + m)*E_ + e0] = o;
  }
}

extern "C" void kernel_launch(void* const* d_in, const int* in_sizes, int n_in,
                              void* d_out, int out_size, void* d_ws, size_t ws_size,
                              hipStream_t stream) {
  (void)in_sizes; (void)n_in; (void)out_size; (void)ws_size;
  const float* x      = (const float*)d_in[0];
  const float* masker = (const float*)d_in[1];
  const float* ln_g   = (const float*)d_in[2];
  const float* ln_b   = (const float*)d_in[3];
  const float* w_t    = (const float*)d_in[4];
  const float* b_t    = (const float*)d_in[5];
  const float* bn_g   = (const float*)d_in[6];
  const float* bn_b   = (const float*)d_in[7];
  const float* bn_m   = (const float*)d_in[8];
  const float* bn_v   = (const float*)d_in[9];
  const float* gate_w = (const float*)d_in[10];
  const float* gate_b = (const float*)d_in[11];
  float* out = (float*)d_out;

  char* ws = (char*)d_ws;
  size_t o = 0;
  half_t* adjT = (half_t*)(ws + o); o += (size_t)T_*FP*KG   * sizeof(half_t);
  half_t* wT16 = (half_t*)(ws + o); o += (size_t)T_*E_*K2E  * sizeof(half_t);
  half_t* gwT16= (half_t*)(ws + o); o += (size_t)FP*GK      * sizeof(half_t);
  float2* bnss = (float2*)(ws + o); o += (size_t)NBN        * sizeof(float2);
  half_t* xcur = (half_t*)(ws + o); // B*T*F*E f16 staging (~42 MB)

  prep_kernel<<<64, 256, 0, stream>>>(masker, ln_g, ln_b, w_t, gate_w,
                                      bn_g, bn_b, bn_m, bn_v,
                                      adjT, wT16, gwT16, bnss);
  main_kernel<<<(B_/2)*T_, 256, 0, stream>>>(x, b_t, adjT, wT16, bnss, xcur);
  gate_kernel<<<(B_*T_)/16, 128, 0, stream>>>(xcur, gwT16, gate_b, out);
}
